// LSTM_20804821582406
// MI455X (gfx1250) — compile-verified, matrix operand order checked by reading, not run on hardware
//
#include <hip/hip_runtime.h>
#include <hip/hip_bf16.h>

// Problem dims (fixed by reference)
#define B_ 256
#define T_ 1024
#define I_ 128
#define H_ 256

typedef __attribute__((ext_vector_type(16))) __bf16 v16bf;
typedef __attribute__((ext_vector_type(8)))  float  v8f;

// Packed weight sizes in ushort (bf16 bit) elements.
// Wh: 4 gates x 16 ntiles x 8 ksteps x (32 lanes x 16 halves)
#define WH_ELEMS (4 * 16 * 8 * 32 * 16)   // 262144  (512 KB)
// Wx: 4 gates x 16 ntiles x 4 ksteps x (32 lanes x 16 halves)
#define WX_ELEMS (4 * 16 * 4 * 32 * 16)   // 131072  (256 KB)

__device__ __forceinline__ unsigned short f2bf(float f) {
    unsigned u = __float_as_uint(f);
    u += 0x7FFFu + ((u >> 16) & 1u);      // round-to-nearest-even
    return (unsigned short)(u >> 16);
}

// K index within a 32-wide K block for (lane, half j) per the documented
// 16-bit matrix fragment layout: two contiguous 8-element runs per lane.
__device__ __forceinline__ int koff(int lane, int j) {
    int off = (lane >= 16) ? 8 : 0;
    return (j < 8) ? (off + j) : (16 + off + (j - 8));
}

__device__ __forceinline__ float sigmoidf(float v) {
    return 1.0f / (1.0f + __expf(-v));
}
__device__ __forceinline__ float tanh_fast(float v) {
    return 1.0f - 2.0f / (__expf(2.0f * v) + 1.0f);
}

__device__ __forceinline__ v8f splat8(float v) {
    v8f r;
    #pragma unroll
    for (int i = 0; i < 8; ++i) r[i] = v;
    return r;
}

struct WPtrs { const float* wh[4]; const float* wx[4]; };

// ---------------------------------------------------------------------------
// Prep: convert f32 weights -> bf16, pre-swizzled into WMMA B-fragment order.
// Element order: [gate][ntile][kstep][lane][half]; each lane's 16 halves are
// contiguous (32 bytes) so the compute kernel loads a fragment as 2x b128.
// ---------------------------------------------------------------------------
__global__ void pack_weights(WPtrs p, unsigned short* __restrict__ dst) {
    int idx = blockIdx.x * blockDim.x + threadIdx.x;
    if (idx < WH_ELEMS) {
        int j    = idx & 15;
        int lane = (idx >> 4) & 31;
        int ks   = (idx >> 9) & 7;
        int nt   = (idx >> 12) & 15;
        int g    = idx >> 16;
        int k = ks * 32 + koff(lane, j);
        int n = nt * 16 + (lane & 15);
        dst[idx] = f2bf(p.wh[g][k * H_ + n]);
    } else if (idx < WH_ELEMS + WX_ELEMS) {
        int r    = idx - WH_ELEMS;
        int j    = r & 15;
        int lane = (r >> 4) & 31;
        int ks   = (r >> 9) & 3;
        int nt   = (r >> 11) & 15;
        int g    = r >> 15;
        int k = ks * 32 + koff(lane, j);
        int n = nt * 16 + (lane & 15);
        dst[idx] = f2bf(p.wx[g][k * H_ + n]);
    }
}

// ---------------------------------------------------------------------------
// Recurrent scan. One workgroup per 16-row batch tile (16 WGs total).
// 16 waves (512 threads); wave w owns ntile w x all 4 gates.
// short state: bf16 in LDS [16][256]; long state: f32 C-fragments in regs.
// xt is cooperatively staged to LDS as bf16 once per step (each thread
// converts 4 floats). Weight B-fragments stream from L2 every step; a
// laundered zero offset stops LLVM hoisting them (would spill: ~384 VGPRs of weights)
// while keeping the loads on the global_load SADDR path (not FLAT).
// ---------------------------------------------------------------------------
__global__ void __launch_bounds__(512, 1)
lstm_scan(const float* __restrict__ x,
          const float* __restrict__ b_f, const float* __restrict__ b_perc,
          const float* __restrict__ b_pot, const float* __restrict__ b_o,
          const unsigned short* __restrict__ wpack,
          float* __restrict__ out)
{
    __shared__ unsigned short s_short[16 * H_];   // bf16 bits, row-major
    __shared__ unsigned short s_x[16 * I_];       // bf16 xt tile, row-major

    const int tid    = threadIdx.x;
    const int lane   = tid & 31;
    const int nt     = tid >> 5;                  // wave id == ntile (0..15)
    const int bb     = blockIdx.x;                // batch tile
    const int laneN  = lane & 15;
    const int hiHalf = (lane >= 16) ? 1 : 0;
    const int off    = hiHalf * 8;

    // zero initial short state
    for (int i = tid; i < 16 * H_; i += 512) s_short[i] = 0;

    // per-gate bias value (depends only on N column of the C fragment)
    const float* bias[4] = { b_f, b_perc, b_pot, b_o };
    float bv[4];
    #pragma unroll
    for (int g = 0; g < 4; ++g) bv[g] = bias[g][nt * 16 + laneN];

    v8f lng   = splat8(0.0f);
    v8f sfrag = splat8(0.0f);

    const int M = laneN;                          // fragment row for A
    // staging source: wave nt stages row nt of the x tile, 4 floats per lane
    const float* xstage = x + ((size_t)(bb * 16 + nt) * T_) * I_ + lane * 4;

    __syncthreads();

    for (int t = 0; t < T_; ++t) {
        // Laundered zero offset: keeps weight loads un-hoistable across the
        // t-loop but rooted at the global kernarg pointer (global_load form).
        size_t lz = 0;
        asm volatile("" : "+s"(lz));
        const unsigned short* whp = wpack + lz;
        const unsigned short* wxp = wpack + (WH_ELEMS + lz);

        // ---- cooperative stage of xt -> LDS bf16 (each thread: 4 floats) --
        {
            float4 xv = *(const float4*)(xstage + (size_t)t * I_);
            unsigned lo = (unsigned)f2bf(xv.x) | ((unsigned)f2bf(xv.y) << 16);
            unsigned hi = (unsigned)f2bf(xv.z) | ((unsigned)f2bf(xv.w) << 16);
            *(uint2*)(s_x + nt * I_ + lane * 4) = make_uint2(lo, hi);
        }

        // ---- A fragments for the hidden part (old short, from LDS) --------
        v16bf a_h[8];
        {
            const unsigned short* row = s_short + M * H_;
            #pragma unroll
            for (int ks = 0; ks < 8; ++ks) {
                uint4* aq = (uint4*)&a_h[ks];
                aq[0] = *(const uint4*)(row + ks * 32 + off);
                aq[1] = *(const uint4*)(row + ks * 32 + 16 + off);
            }
        }
        __syncthreads();   // staging done; old-short reads done

        // ---- A fragments for the input part (xt, from LDS) ----------------
        v16bf a_x[4];
        {
            const unsigned short* row = s_x + M * I_;
            #pragma unroll
            for (int ks = 0; ks < 4; ++ks) {
                uint4* aq = (uint4*)&a_x[ks];
                aq[0] = *(const uint4*)(row + ks * 32 + off);
                aq[1] = *(const uint4*)(row + ks * 32 + 16 + off);
            }
        }

        v8f acc[4];
        #pragma unroll
        for (int g = 0; g < 4; ++g) {
            acc[g] = splat8(0.0f);   // bias added in the elementwise phase
            // hidden part: K = 256 -> 8 k-steps
            #pragma unroll
            for (int ks = 0; ks < 8; ++ks) {
                const unsigned eoff =
                    (unsigned)(((g * 16 + nt) * 8 + ks) * 512 + lane * 16);
                v16bf bfrag;
                uint4* bq = (uint4*)&bfrag;
                bq[0] = *(const uint4*)(whp + eoff);
                bq[1] = *(const uint4*)(whp + eoff + 8);
                acc[g] = __builtin_amdgcn_wmma_f32_16x16x32_bf16(
                    false, a_h[ks], false, bfrag, (short)0, acc[g], false, false);
            }
            // input part: K = 128 -> 4 k-steps
            #pragma unroll
            for (int ks = 0; ks < 4; ++ks) {
                const unsigned eoff =
                    (unsigned)(((g * 16 + nt) * 4 + ks) * 512 + lane * 16);
                v16bf bfrag;
                uint4* bq = (uint4*)&bfrag;
                bq[0] = *(const uint4*)(wxp + eoff);
                bq[1] = *(const uint4*)(wxp + eoff + 8);
                acc[g] = __builtin_amdgcn_wmma_f32_16x16x32_bf16(
                    false, a_x[ks], false, bfrag, (short)0, acc[g], false, false);
            }
        }

        // ---- fused gate nonlinearity on f32 C fragments (bias added here) -
        #pragma unroll
        for (int i = 0; i < 8; ++i) {
            float fg = sigmoidf(acc[0][i] + bv[0]);
            float pg = sigmoidf(acc[1][i] + bv[1]);
            float tg = tanh_fast(acc[2][i] + bv[2]);
            float og = sigmoidf(acc[3][i] + bv[3]);
            float nl = fg * lng[i] + pg * tg;
            lng[i]   = nl;
            sfrag[i] = tanh_fast(nl) * og;
        }

        // write new short to LDS (bf16) per C layout: row = v + hiHalf*8
        #pragma unroll
        for (int v = 0; v < 8; ++v) {
            int Mr = v + hiHalf * 8;
            s_short[Mr * H_ + nt * 16 + laneN] = f2bf(sfrag[v]);
        }
        __syncthreads();   // new short visible to all waves next step
    }

    // ---- final outputs: out[0 : B*H) = short, out[B*H : 2*B*H) = long ----
    #pragma unroll
    for (int v = 0; v < 8; ++v) {
        int brow = bb * 16 + v + hiHalf * 8;
        int col  = nt * 16 + laneN;
        out[(size_t)brow * H_ + col]                   = sfrag[v];
        out[(size_t)B_ * H_ + (size_t)brow * H_ + col] = lng[v];
    }
}

extern "C" void kernel_launch(void* const* d_in, const int* in_sizes, int n_in,
                              void* d_out, int out_size, void* d_ws, size_t ws_size,
                              hipStream_t stream) {
    (void)in_sizes; (void)n_in; (void)out_size; (void)ws_size;
    const float* x = (const float*)d_in[0];
    WPtrs p;
    p.wh[0] = (const float*)d_in[1];  p.wx[0] = (const float*)d_in[2];
    const float* b_f    = (const float*)d_in[3];
    p.wh[1] = (const float*)d_in[4];  p.wx[1] = (const float*)d_in[5];
    const float* b_perc = (const float*)d_in[6];
    p.wh[2] = (const float*)d_in[7];  p.wx[2] = (const float*)d_in[8];
    const float* b_pot  = (const float*)d_in[9];
    p.wh[3] = (const float*)d_in[10]; p.wx[3] = (const float*)d_in[11];
    const float* b_o    = (const float*)d_in[12];

    unsigned short* wpack = (unsigned short*)d_ws;   // 768 KB used

    const int total = WH_ELEMS + WX_ELEMS;
    pack_weights<<<(total + 255) / 256, 256, 0, stream>>>(p, wpack);
    lstm_scan<<<16, 512, 0, stream>>>(x, b_f, b_perc, b_pot, b_o, wpack,
                                      (float*)d_out);
}
